// WavLMGumbelVectorQuantizer_46385646797464
// MI455X (gfx1250) — matrix operand order repaired and checked by reading, not compiled, not run on hardware
//
#include <hip/hip_runtime.h>
#include <hip/hip_bf16.h>

typedef __attribute__((ext_vector_type(16))) __bf16 bf16x16;
typedef __attribute__((ext_vector_type(8)))  float  f32x8;
typedef unsigned short ushort_t;
typedef unsigned int   uint_t;

#define NFRAMES   65536
#define HDIM      512
#define GVDIM     640
#define VNUM      320
#define DDIM      128
#define ROWS      32
#define NKB       16
// LDS byte offsets (GEMM phase). A/W rows padded to 40 bf16 (80 B) for bank spread.
#define AH_OFF    0          // 32 rows * 80 B
#define AL_OFF    2560
#define WH_OFF    5120       // 640 rows * 80 B (transposed: [n][k])
#define WL_OFF    56320
#define LDS_BYTES 107520
// LDS byte offsets (softmax phase, reuses the above after a barrier)
#define LOGSTR    644        // dwords; 8*644 % 64 == 32 -> conflict-free tile writes
#define MARG_OFF  82432      // = 32*LOGSTR*4

__device__ __forceinline__ uint_t f2bfbits(float x) {
    uint_t u = __float_as_uint(x);
    u += 0x7FFFu + ((u >> 16) & 1u);   // round-to-nearest-even
    return u >> 16;
}
__device__ __forceinline__ float bfbits2f(uint_t s) {
    return __uint_as_float(s << 16);
}

union FragU { bf16x16 v; uint4 q[2]; };

__global__ __launch_bounds__(256) void vq_main(
    const float* __restrict__ hidden, const float* __restrict__ W,
    const float* __restrict__ bias,   const float* __restrict__ codevectors,
    const float* __restrict__ gumbel, float* __restrict__ out,
    float* __restrict__ marg_glob)
{
    extern __shared__ __align__(16) char smem[];
    const int tid  = threadIdx.x;
    const int lane = tid & 31;
    const int w    = tid >> 5;
    const int rb   = w & 1;          // row block (16 rows)
    const int cg   = w >> 1;         // col group (160 cols)
    const int nl   = lane & 15;
    const int hi   = lane >> 4;
    const long row0 = (long)blockIdx.x * ROWS;

    ushort_t* ah_s = (ushort_t*)(smem + AH_OFF);
    ushort_t* al_s = (ushort_t*)(smem + AL_OFF);
    ushort_t* wh_s = (ushort_t*)(smem + WH_OFF);
    ushort_t* wl_s = (ushort_t*)(smem + WL_OFF);

    f32x8 acc[10] = {};

    for (int kb = 0; kb < NKB; ++kb) {
        if (kb) __syncthreads();     // previous compute done before overwrite
        // ---- stage A slice: 32 rows x 32 k, split into bf16 hi/lo ----
        {
            const int r  = tid >> 3;
            const int h4 = (tid & 7) << 2;
            float4 v = *(const float4*)(hidden + (row0 + r) * HDIM + kb * 32 + h4);
            float xs[4] = { v.x, v.y, v.z, v.w };
            uint_t hb[4], lb[4];
            #pragma unroll
            for (int c = 0; c < 4; ++c) {
                hb[c] = f2bfbits(xs[c]);
                lb[c] = f2bfbits(xs[c] - bfbits2f(hb[c]));
            }
            const int eo = r * 40 + h4;  // element offset (row stride 40 bf16)
            *(uint_t*)((char*)ah_s + eo * 2)     = hb[0] | (hb[1] << 16);
            *(uint_t*)((char*)ah_s + eo * 2 + 4) = hb[2] | (hb[3] << 16);
            *(uint_t*)((char*)al_s + eo * 2)     = lb[0] | (lb[1] << 16);
            *(uint_t*)((char*)al_s + eo * 2 + 4) = lb[2] | (lb[3] << 16);
        }
        // ---- stage W slice: 32 k x 640 n, transposed into LDS as [n][k] ----
        for (int i = tid; i < 32 * 160; i += 256) {
            const int k  = i / 160;
            const int n4 = (i % 160) << 2;
            float4 v = *(const float4*)(W + (size_t)(kb * 32 + k) * GVDIM + n4);
            float xs[4] = { v.x, v.y, v.z, v.w };
            #pragma unroll
            for (int c = 0; c < 4; ++c) {
                uint_t hb = f2bfbits(xs[c]);
                uint_t lb = f2bfbits(xs[c] - bfbits2f(hb));
                wh_s[(n4 + c) * 40 + k] = (ushort_t)hb;
                wl_s[(n4 + c) * 40 + k] = (ushort_t)lb;
            }
        }
        __syncthreads();
        // ---- A fragments (16x32 bf16, ISA layout: lane<16 K in {8hi..8hi+7, 16+8hi..}) ----
        FragU Ahf, Alf;
        {
            const char* ba = (const char*)ah_s + ((rb * 16 + nl) * 40 + 8 * hi) * 2;
            Ahf.q[0] = *(const uint4*)ba;
            Ahf.q[1] = *(const uint4*)(ba + 32);
            const char* bl_ = (const char*)al_s + ((rb * 16 + nl) * 40 + 8 * hi) * 2;
            Alf.q[0] = *(const uint4*)bl_;
            Alf.q[1] = *(const uint4*)(bl_ + 32);
        }
        #pragma unroll
        for (int ct = 0; ct < 10; ++ct) {
            const int n = cg * 160 + ct * 16 + nl;   // B: lane -> N, K in {8hi..8hi+7, 16+8hi..}
            const char* bh = (const char*)wh_s + (n * 40 + 8 * hi) * 2;
            const char* bl = (const char*)wl_s + (n * 40 + 8 * hi) * 2;
            FragU Bh, Bl;
            Bh.q[0] = *(const uint4*)bh;  Bh.q[1] = *(const uint4*)(bh + 32);
            Bl.q[0] = *(const uint4*)bl;  Bl.q[1] = *(const uint4*)(bl + 32);
            acc[ct] = __builtin_amdgcn_wmma_f32_16x16x32_bf16(false, Ahf.v, false, Bh.v, (short)0, acc[ct], false, false);
            acc[ct] = __builtin_amdgcn_wmma_f32_16x16x32_bf16(false, Ahf.v, false, Bl.v, (short)0, acc[ct], false, false);
            acc[ct] = __builtin_amdgcn_wmma_f32_16x16x32_bf16(false, Alf.v, false, Bh.v, (short)0, acc[ct], false, false);
        }
    }
    __syncthreads();

    // ---- phase 2: bias add, spill logits to LDS (C layout: m = 8*hi+i, n = lane&15) ----
    float* logits = (float*)(smem);
    float* marg   = (float*)(smem + MARG_OFF);
    #pragma unroll
    for (int ct = 0; ct < 10; ++ct) {
        const int n = cg * 160 + ct * 16 + nl;
        const float bv = bias[n];
        #pragma unroll
        for (int i = 0; i < 8; ++i) {
            const int m = rb * 16 + hi * 8 + i;
            logits[m * LOGSTR + n] = acc[ct][i] + bv;
        }
    }
    for (int i = tid; i < GVDIM; i += 256) marg[i] = 0.f;
    __syncthreads();

    // ---- phase 3: per (group, row) softmaxes; one wave owns (grp, 8 rows) ----
    const int grp   = w & 1;
    const int rbase = (w >> 1) * 8;
    float macc[10];
    #pragma unroll
    for (int t = 0; t < 10; ++t) macc[t] = 0.f;

    for (int q = 0; q < 8; ++q) {
        const int row = rbase + q;
        const long nglob = row0 + row;
        float l[10];
        #pragma unroll
        for (int t = 0; t < 10; ++t)
            l[t] = logits[row * LOGSTR + grp * VNUM + lane + 32 * t];

        // noise-free softmax -> marginal accumulation
        float mx = l[0];
        #pragma unroll
        for (int t = 1; t < 10; ++t) mx = fmaxf(mx, l[t]);
        #pragma unroll
        for (int m = 16; m > 0; m >>= 1) mx = fmaxf(mx, __shfl_xor(mx, m, 32));
        float e[10], s = 0.f;
        #pragma unroll
        for (int t = 0; t < 10; ++t) { e[t] = expf(l[t] - mx); s += e[t]; }
        #pragma unroll
        for (int m = 16; m > 0; m >>= 1) s += __shfl_xor(s, m, 32);
        const float inv = 1.f / s;
        #pragma unroll
        for (int t = 0; t < 10; ++t) macc[t] += e[t] * inv;

        // gumbel-noisy softmax: argmax + straight-through coefficient
        const float* gu = gumbel + ((size_t)nglob * 2 + grp) * VNUM + lane;
        float tv[10], best = -3.4e38f;
        int bidx = 0;
        #pragma unroll
        for (int t = 0; t < 10; ++t) {
            const float u  = gu[32 * t];
            const float gn = -logf(-logf(u));
            tv[t] = (l[t] + gn) * 0.5f;                 // TAU = 2
            if (tv[t] > best) { best = tv[t]; bidx = lane + 32 * t; }
        }
        #pragma unroll
        for (int m = 16; m > 0; m >>= 1) {
            const float ob = __shfl_xor(best, m, 32);
            const int   oi = __shfl_xor(bidx, m, 32);
            if (ob > best || (ob == best && oi < bidx)) { best = ob; bidx = oi; }
        }
        float s2 = 0.f;
        #pragma unroll
        for (int t = 0; t < 10; ++t) s2 += expf(tv[t] - best);
        #pragma unroll
        for (int m = 16; m > 0; m >>= 1) s2 += __shfl_xor(s2, m, 32);
        const float smax = 1.f / s2;                    // y_soft at argmax
        const float coef = (1.f - smax) + smax;         // matches y_hard - sg(y_soft) + y_soft

        const float4 c4 = ((const float4*)(codevectors + ((size_t)(grp * VNUM + bidx)) * DDIM))[lane];
        float4 o; o.x = coef * c4.x; o.y = coef * c4.y; o.z = coef * c4.z; o.w = coef * c4.w;
        ((float4*)(out + (size_t)nglob * (2 * DDIM) + grp * DDIM))[lane] = o;
    }
    #pragma unroll
    for (int t = 0; t < 10; ++t)
        atomicAdd(&marg[grp * VNUM + lane + 32 * t], macc[t]);
    __syncthreads();
    for (int i = tid; i < GVDIM; i += 256)
        atomicAdd(&marg_glob[i], marg[i]);
}

__global__ void vq_zero(float* __restrict__ ws) { ws[threadIdx.x] = 0.f; }

__global__ void vq_finalize(const float* __restrict__ ws, float* __restrict__ out_perp) {
    __shared__ float part[2];
    const int tid = threadIdx.x;
    if (tid < 2) part[tid] = 0.f;
    __syncthreads();
    const float m = ws[tid] * (1.0f / (float)NFRAMES);
    const float term = m * logf(m + 1e-7f);
    atomicAdd(&part[tid / VNUM], term);
    __syncthreads();
    if (tid == 0) out_perp[0] = expf(-part[0]) + expf(-part[1]);
}

extern "C" void kernel_launch(void* const* d_in, const int* in_sizes, int n_in,
                              void* d_out, int out_size, void* d_ws, size_t ws_size,
                              hipStream_t stream) {
    (void)in_sizes; (void)n_in; (void)out_size; (void)ws_size;
    const float* hidden = (const float*)d_in[0];
    const float* W      = (const float*)d_in[1];
    const float* b      = (const float*)d_in[2];
    const float* cv     = (const float*)d_in[3];
    const float* gu     = (const float*)d_in[4];
    float* out  = (float*)d_out;
    float* marg = (float*)d_ws;   // 640 floats of scratch

    vq_zero<<<1, GVDIM, 0, stream>>>(marg);
    vq_main<<<NFRAMES / ROWS, 256, LDS_BYTES, stream>>>(hidden, W, b, cv, gu, out, marg);
    vq_finalize<<<1, GVDIM, 0, stream>>>(marg, out + (size_t)NFRAMES * 2 * DDIM);
}
